// GAT_Batch_19481971655237
// MI455X (gfx1250) — matrix operand order repaired and checked by reading, not compiled, not run on hardware
//
#include <hip/hip_runtime.h>
#include <hip/hip_bf16.h>
#include <stdint.h>

// ---------------------------------------------------------------------------
// GAT batch forward for gfx1250 (CDNA5, wave32, WMMA 16x16x32 bf16).
// B=32, N=512, F_in=128, F_hid=64, H=8, nclass=16.
//
// All matmul "B" operands are kept K-major (transposed) so B-fragments are
// two contiguous b128 loads; softmax numerators live in LDS as packed bf16
// pairs with a 4-DWORD row skew; normalization is folded into the WMMA
// epilogue; adj mask is cached in a per-thread 64-bit bitfield.
// k2 prefetches its B slab (global_prefetch_b8); k4 async-stages its B slab
// into LDS (global_load_async_to_lds_b128 / s_wait_asynccnt).
// ---------------------------------------------------------------------------

typedef __bf16 bf16_t;
typedef __attribute__((ext_vector_type(16))) __bf16 v16bf;
typedef __attribute__((ext_vector_type(8)))  __bf16 v8bf;
typedef __attribute__((ext_vector_type(8)))  float  v8f;
typedef __attribute__((ext_vector_type(4)))  float  v4f;
typedef __attribute__((ext_vector_type(2)))  float  v2f;
typedef __attribute__((ext_vector_type(2)))  int    v2i;

#define B_   32
#define N_   512
#define FIN_ 128
#define FH_  64
#define H_   8
#define NC_  16
#define HF_  (H_ * FH_)          // 512
#define ALPHA_   0.2f
#define NEG_INF_ (-9.0e15f)
#define ATT_STRIDE 260           // DWORDs per softmax row (256 pairs + 4 skew)
#define BS_STRIDE  520           // bf16 elems per staged B row (512 + 8 skew)

union BF2 { uint32_t u; __bf16 h[2]; };

static __device__ __forceinline__ v8f wmma_bf16(v16bf a, v16bf b, v8f c) {
  return __builtin_amdgcn_wmma_f32_16x16x32_bf16(
      false, a, false, b, (short)0, c, false, false);
}

static __device__ __forceinline__ float gat_score(float f1v, float f2v, int m) {
  float s = f1v + f2v;
  s = s > 0.f ? s : ALPHA_ * s;          // LeakyReLU(0.2)
  return m ? s : NEG_INF_;               // adjacency mask
}

// Fragment maps (wave32):
//  A elem e(0..7)->K=ka+e, e(8..15)->K=ka+16+e-8, ka=k0+8*(lane>=16), row=lane&15
//  B elem e(0..7)->K=kb+e, e(8..15)->K=kb+16+e-8, kb=k0+8*(lane>=16), col=lane&15
//  C/D elem v(0..7)->M=v+8*(lane>=16), N=lane&15

// ---------------------------------------------------------------------------
// Kernel 0: pack weights to K-major bf16.
//   WhT[h][c][k] = W_heads[h][k][c]   (8 x 64 x 128)
//   WoT[c][k]    = W_out[k][c]        (16 x 512)
// ---------------------------------------------------------------------------
__global__ __launch_bounds__(256) void k0_pack(
    const float* __restrict__ Wh, const float* __restrict__ Wo,
    bf16_t* __restrict__ WhT, bf16_t* __restrict__ WoT)
{
  const int idx = blockIdx.x * 256 + threadIdx.x;        // 0 .. 65535
  if (idx < H_ * FH_ * FIN_) {
    const int h = idx / (FH_ * FIN_);
    const int r = idx % (FH_ * FIN_);
    const int c = r / FIN_, k = r % FIN_;
    WhT[idx] = (bf16_t)Wh[((size_t)h * FIN_ + k) * FH_ + c];
  }
  if (idx < NC_ * HF_) {
    const int c = idx / HF_, k = idx % HF_;
    WoT[idx] = (bf16_t)Wo[(size_t)k * NC_ + c];
  }
}

// ---------------------------------------------------------------------------
// Kernel 1: hfeatT[h][b][c][n] = (x[b,n,:] @ W[h])^T, fused f1/f2 projections.
// 4 waves; wave owns 16 output cols of a 16-row tile. Epilogue stores one
// contiguous b128 (8 rows of one col) per lane.
// ---------------------------------------------------------------------------
__global__ __launch_bounds__(128) void k1_head_gemm(
    const float* __restrict__ x,      // [B,N,128]
    const bf16_t* __restrict__ WhT,   // [H,64,128]
    const float* __restrict__ a1,     // [H,64]
    const float* __restrict__ a2,     // [H,64]
    bf16_t* __restrict__ hfeatT,      // [H,B,64,512]
    float* __restrict__ f1,           // [H,B,N]
    float* __restrict__ f2)           // [H,B,N]
{
  const int h  = blockIdx.y;
  const int b  = blockIdx.x >> 5;
  const int m0 = (blockIdx.x & 31) << 4;
  const int wave = threadIdx.x >> 5, lane = threadIdx.x & 31;
  const int lo = lane & 15, hi = lane >> 4;

  __shared__ float s1[16], s2[16];
  if (threadIdx.x < 16) { s1[threadIdx.x] = 0.f; s2[threadIdx.x] = 0.f; }
  __syncthreads();

  const float*  xrow = x + ((size_t)b * N_ + (m0 + lo)) * FIN_;
  const int     colg = wave * 16 + lo;
  const bf16_t* wrow = WhT + ((size_t)h * FH_ + colg) * FIN_;

  v8f acc = {0.f, 0.f, 0.f, 0.f, 0.f, 0.f, 0.f, 0.f};
#pragma unroll
  for (int k0 = 0; k0 < FIN_; k0 += 32) {
    const int ka = k0 + hi * 8;
    v4f a0 = *(const v4f*)(xrow + ka);
    v4f a1q = *(const v4f*)(xrow + ka + 4);
    v4f a2q = *(const v4f*)(xrow + ka + 16);
    v4f a3q = *(const v4f*)(xrow + ka + 20);
    const int kb = k0 + hi * 8;
    v8bf b0 = *(const v8bf*)(wrow + kb);
    v8bf b1 = *(const v8bf*)(wrow + kb + 16);
    v16bf af, bfr;
#pragma unroll
    for (int e = 0; e < 4; ++e) {
      af[e]      = (bf16_t)a0[e];
      af[e + 4]  = (bf16_t)a1q[e];
      af[e + 8]  = (bf16_t)a2q[e];
      af[e + 12] = (bf16_t)a3q[e];
    }
#pragma unroll
    for (int e = 0; e < 8; ++e) { bfr[e] = b0[e]; bfr[e + 8] = b1[e]; }
    acc = wmma_bf16(af, bfr, acc);
  }

  const float a1v = a1[h * FH_ + colg];
  const float a2v = a2[h * FH_ + colg];
  v8bf st;
#pragma unroll
  for (int v = 0; v < 8; ++v) {
    const int M = hi * 8 + v;
    const float val = acc[v];
    st[v] = (bf16_t)val;
    atomicAdd(&s1[M], val * a1v);                 // ds_add_f32
    atomicAdd(&s2[M], val * a2v);
  }
  // contiguous store: 8 rows (n = m0+hi*8 .. +7) of column colg
  *(v8bf*)(hfeatT + (((size_t)h * B_ + b) * FH_ + colg) * N_ + m0 + hi * 8) = st;

  __syncthreads();
  if (threadIdx.x < 16) {
    const size_t idx = ((size_t)h * B_ + b) * N_ + m0 + threadIdx.x;
    f1[idx] = s1[threadIdx.x];
    f2[idx] = s2[threadIdx.x];
  }
}

// ---------------------------------------------------------------------------
// Kernel 2: fused masked softmax + aggregation + ELU per head.
// Block = 16 query rows of one (h,b). B slab (64x512 bf16, K-major) prefetched
// while softmax runs; numerators stored as packed bf16 pairs; 1/sum folded
// into the WMMA epilogue.
// ---------------------------------------------------------------------------
__global__ __launch_bounds__(128) void k2_head_attn(
    const int* __restrict__ adj,       // [B,N,N]
    const bf16_t* __restrict__ hfeatT, // [H,B,64,512]
    const float* __restrict__ f1,      // [H,B,N]
    const float* __restrict__ f2,      // [H,B,N]
    bf16_t* __restrict__ xcat)         // [B,N,512]
{
  const int h  = blockIdx.y;
  const int b  = blockIdx.x >> 5;
  const int i0 = (blockIdx.x & 31) << 4;
  const int tid = threadIdx.x;

  __shared__ __align__(16) uint32_t s_att[16 * ATT_STRIDE];  // 16.6 KB
  __shared__ __align__(16) float    s_f2[N_];                // 2 KB
  __shared__ float s_red[16][8];
  __shared__ float s_inv[16];

  const size_t hboff = ((size_t)h * B_ + b) * N_;
  const bf16_t* gB = hfeatT + hboff * FH_;   // [64][512] K-major slab

  // prefetch the 64 KB B slab into cache while we do softmax (512 B/thread)
  {
    const char* p = (const char*)gB + (size_t)tid * 512;
#pragma unroll
    for (int i = 0; i < 4; ++i) __builtin_prefetch(p + i * 128, 0, 1);
  }
  // stage f2 row (coalesced)
  *(v4f*)&s_f2[tid * 4] = *(const v4f*)(f2 + hboff + tid * 4);
  __syncthreads();

  const int row = tid >> 3, sub = tid & 7;   // 8 threads per query row
  const float f1v = f1[hboff + i0 + row];
  const int* adjr = adj + ((size_t)b * N_ + (i0 + row)) * N_;

  // pass 1: row max; cache adjacency in a 64-bit bitfield (adj read once)
  uint64_t mbits = 0;
  float mx = -3.0e38f;
  for (int t = 0; t < 32; ++t) {
    const int q = sub + 8 * t;               // pair index 0..255
    v2i m  = *(const v2i*)(adjr + 2 * q);
    v2f fv = *(const v2f*)&s_f2[2 * q];
    const int m0i = m[0] > 0, m1i = m[1] > 0;
    mbits |= ((uint64_t)(m0i | (m1i << 1))) << (2 * t);
    const float sc0 = gat_score(f1v, fv[0], m0i);
    const float sc1 = gat_score(f1v, fv[1], m1i);
    mx = fmaxf(mx, fmaxf(sc0, sc1));
  }
  s_red[row][sub] = mx;
  __syncthreads();
  mx = s_red[row][0];
#pragma unroll
  for (int t = 1; t < 8; ++t) mx = fmaxf(mx, s_red[row][t]);
  __syncthreads();

  // pass 2: numerators -> packed bf16 pairs in LDS, accumulate row sum
  float sum = 0.f;
  for (int t = 0; t < 32; ++t) {
    const int q = sub + 8 * t;
    v2f fv = *(const v2f*)&s_f2[2 * q];
    const int m0i = (int)((mbits >> (2 * t)) & 1);
    const int m1i = (int)((mbits >> (2 * t + 1)) & 1);
    const float p0 = __expf(gat_score(f1v, fv[0], m0i) - mx);
    const float p1 = __expf(gat_score(f1v, fv[1], m1i) - mx);
    BF2 pk; pk.h[0] = (bf16_t)p0; pk.h[1] = (bf16_t)p1;
    s_att[row * ATT_STRIDE + q] = pk.u;      // skewed: conflict-minimal
    sum += (float)pk.h[0] + (float)pk.h[1];
  }
  s_red[row][sub] = sum;
  __syncthreads();
  float tot = 0.f;
#pragma unroll
  for (int t = 0; t < 8; ++t) tot += s_red[row][t];
  if (sub == 0) s_inv[row] = 1.f / tot;
  __syncthreads();

  // out(16x64) = p(16x512) @ hfeat(512x64), scaled per-row by 1/sum
  const int wave = tid >> 5, lane = tid & 31, lo = lane & 15, hi = lane >> 4;
  const int colg = wave * 16 + lo;
  const bf16_t* arow = (const bf16_t*)&s_att[lo * ATT_STRIDE];
  const bf16_t* brow = gB + (size_t)colg * N_;

  v8f acc = {0.f, 0.f, 0.f, 0.f, 0.f, 0.f, 0.f, 0.f};
  for (int k0 = 0; k0 < N_; k0 += 32) {
    const int ka = k0 + hi * 8;
    v8bf aLo = *(const v8bf*)(arow + ka);        // ds_load_b128, skewed
    v8bf aHi = *(const v8bf*)(arow + ka + 16);
    const int kb = k0 + hi * 8;
    v8bf b0 = *(const v8bf*)(brow + kb);         // contiguous global b128
    v8bf b1 = *(const v8bf*)(brow + kb + 16);
    v16bf af, bfr;
#pragma unroll
    for (int e = 0; e < 8; ++e) {
      af[e] = aLo[e]; af[e + 8] = aHi[e];
      bfr[e] = b0[e]; bfr[e + 8] = b1[e];
    }
    acc = wmma_bf16(af, bfr, acc);
  }

  bf16_t* xc = xcat + ((size_t)b * N_ + i0) * HF_ + h * FH_ + colg;
#pragma unroll
  for (int v = 0; v < 8; ++v) {
    const int M = hi * 8 + v;
    float val = acc[v] * s_inv[M];
    val = val > 0.f ? val : (__expf(val) - 1.f);   // ELU (concat=True)
    xc[(size_t)M * HF_] = (bf16_t)val;
  }
}

// ---------------------------------------------------------------------------
// Kernel 3: houtT[b][c][n] = (x_cat @ W_out)^T, fused f1o/f2o.
// 4 waves per block, each owns its own 16-row tile (16 cols total).
// ---------------------------------------------------------------------------
__global__ __launch_bounds__(128) void k3_out_gemm(
    const bf16_t* __restrict__ xcat,  // [B,N,512]
    const bf16_t* __restrict__ WoT,   // [16,512]
    const float* __restrict__ a1o,    // [16]
    const float* __restrict__ a2o,    // [16]
    bf16_t* __restrict__ houtT,       // [B,16,512]
    float* __restrict__ f1o,          // [B,N]
    float* __restrict__ f2o)          // [B,N]
{
  const int wave = threadIdx.x >> 5, lane = threadIdx.x & 31;
  const int lo = lane & 15, hi = lane >> 4;
  const int gm = blockIdx.x * 4 + wave;     // global 16-row tile id
  const int b  = gm >> 5;
  const int m0 = (gm & 31) << 4;

  __shared__ float s1[4][16], s2[4][16];
  if (lane < 16) { s1[wave][lane] = 0.f; s2[wave][lane] = 0.f; }
  __syncthreads();

  const bf16_t* arow = xcat + ((size_t)b * N_ + m0 + lo) * HF_;
  const bf16_t* wrow = WoT + (size_t)lo * HF_;

  v8f acc = {0.f, 0.f, 0.f, 0.f, 0.f, 0.f, 0.f, 0.f};
  for (int k0 = 0; k0 < HF_; k0 += 32) {
    const int ka = k0 + hi * 8;
    v8bf aLo = *(const v8bf*)(arow + ka);
    v8bf aHi = *(const v8bf*)(arow + ka + 16);
    const int kb = k0 + hi * 8;
    v8bf b0 = *(const v8bf*)(wrow + kb);
    v8bf b1 = *(const v8bf*)(wrow + kb + 16);
    v16bf af, bfr;
#pragma unroll
    for (int e = 0; e < 8; ++e) {
      af[e] = aLo[e]; af[e + 8] = aHi[e];
      bfr[e] = b0[e]; bfr[e + 8] = b1[e];
    }
    acc = wmma_bf16(af, bfr, acc);
  }

  const float a1v = a1o[lo], a2v = a2o[lo];
  v8bf st;
#pragma unroll
  for (int v = 0; v < 8; ++v) {
    const int M = hi * 8 + v;
    const float val = acc[v];
    st[v] = (bf16_t)val;
    atomicAdd(&s1[wave][M], val * a1v);
    atomicAdd(&s2[wave][M], val * a2v);
  }
  *(v8bf*)(houtT + ((size_t)b * NC_ + lo) * N_ + m0 + hi * 8) = st;

  __syncthreads();
  if (lane < 16) {
    const size_t idx = (size_t)b * N_ + m0 + lane;
    f1o[idx] = s1[wave][lane];
    f2o[idx] = s2[wave][lane];
  }
}

// ---------------------------------------------------------------------------
// Kernel 4: output attention (concat=False), f32 result.
// B slab (16x512 bf16, 16 KB) async-staged into LDS (ASYNCcnt) overlapping the
// softmax; K=512 split across 4 waves; partial C tiles reduced through LDS.
// ---------------------------------------------------------------------------
__global__ __launch_bounds__(128) void k4_out_attn(
    const int* __restrict__ adj,      // [B,N,N]
    const bf16_t* __restrict__ houtT, // [B,16,512]
    const float* __restrict__ f1o,    // [B,N]
    const float* __restrict__ f2o,    // [B,N]
    float* __restrict__ out)          // [B,N,16]
{
  const int b  = blockIdx.x >> 5;
  const int i0 = (blockIdx.x & 31) << 4;
  const int tid = threadIdx.x;

  __shared__ __align__(16) uint32_t s_att[16 * ATT_STRIDE];  // 16.6 KB
  __shared__ __align__(16) bf16_t   s_B[NC_ * BS_STRIDE];    // 16.6 KB
  __shared__ __align__(16) float    s_f2[N_];
  __shared__ float s_red[16][8];
  __shared__ float s_inv[16];
  __shared__ float s_acc[4][16][16];

  // async-stage the 16x512 bf16 B slab into LDS (overlaps softmax below)
  {
    const char* gbase = (const char*)(houtT + (size_t)b * NC_ * N_);
    const uint32_t lbase = (uint32_t)(uintptr_t)&s_B[0];
#pragma unroll
    for (int it = 0; it < 8; ++it) {
      const int q = tid + 128 * it;            // 16-byte chunk id, 0..1023
      const int c = q >> 6, w = q & 63;
      const uint32_t loff = lbase + (uint32_t)(c * (BS_STRIDE * 2) + w * 16);
      const char* g = gbase + (size_t)(c * 1024 + w * 16);
      asm volatile("global_load_async_to_lds_b128 %0, %1, off"
                   :: "v"(loff), "v"(g) : "memory");
    }
  }
  *(v4f*)&s_f2[tid * 4] = *(const v4f*)(f2o + (size_t)b * N_ + tid * 4);
  __syncthreads();

  const int row = tid >> 3, sub = tid & 7;
  const float f1v = f1o[(size_t)b * N_ + i0 + row];
  const int* adjr = adj + ((size_t)b * N_ + (i0 + row)) * N_;

  uint64_t mbits = 0;
  float mx = -3.0e38f;
  for (int t = 0; t < 32; ++t) {
    const int q = sub + 8 * t;
    v2i m  = *(const v2i*)(adjr + 2 * q);
    v2f fv = *(const v2f*)&s_f2[2 * q];
    const int m0i = m[0] > 0, m1i = m[1] > 0;
    mbits |= ((uint64_t)(m0i | (m1i << 1))) << (2 * t);
    mx = fmaxf(mx, fmaxf(gat_score(f1v, fv[0], m0i),
                         gat_score(f1v, fv[1], m1i)));
  }
  s_red[row][sub] = mx;
  __syncthreads();
  mx = s_red[row][0];
#pragma unroll
  for (int t = 1; t < 8; ++t) mx = fmaxf(mx, s_red[row][t]);
  __syncthreads();

  float sum = 0.f;
  for (int t = 0; t < 32; ++t) {
    const int q = sub + 8 * t;
    v2f fv = *(const v2f*)&s_f2[2 * q];
    const int m0i = (int)((mbits >> (2 * t)) & 1);
    const int m1i = (int)((mbits >> (2 * t + 1)) & 1);
    const float p0 = __expf(gat_score(f1v, fv[0], m0i) - mx);
    const float p1 = __expf(gat_score(f1v, fv[1], m1i) - mx);
    BF2 pk; pk.h[0] = (bf16_t)p0; pk.h[1] = (bf16_t)p1;
    s_att[row * ATT_STRIDE + q] = pk.u;
    sum += (float)pk.h[0] + (float)pk.h[1];
  }
  s_red[row][sub] = sum;
  __syncthreads();
  float tot = 0.f;
#pragma unroll
  for (int t = 0; t < 8; ++t) tot += s_red[row][t];
  if (sub == 0) s_inv[row] = 1.f / tot;

  // make sure the async B staging has landed before the matmul
  asm volatile("s_wait_asynccnt 0x0" ::: "memory");
  __syncthreads();

  // split-K: out(16x16) = p(16x512) @ hout(512x16); wave w owns K in [128w,128w+128)
  const int wave = tid >> 5, lane = tid & 31, lo = lane & 15, hi = lane >> 4;
  const bf16_t* arow = (const bf16_t*)&s_att[lo * ATT_STRIDE];
  const bf16_t* brow = &s_B[lo * BS_STRIDE];

  v8f acc = {0.f, 0.f, 0.f, 0.f, 0.f, 0.f, 0.f, 0.f};
  const int kbase = wave * 128;
#pragma unroll
  for (int kk = 0; kk < 4; ++kk) {
    const int k0 = kbase + kk * 32;
    const int ka = k0 + hi * 8;
    v8bf aLo = *(const v8bf*)(arow + ka);
    v8bf aHi = *(const v8bf*)(arow + ka + 16);
    const int kb = k0 + hi * 8;
    v8bf b0 = *(const v8bf*)(brow + kb);
    v8bf b1 = *(const v8bf*)(brow + kb + 16);
    v16bf af, bfr;
#pragma unroll
    for (int e = 0; e < 8; ++e) {
      af[e] = aLo[e]; af[e + 8] = aHi[e];
      bfr[e] = b0[e]; bfr[e + 8] = b1[e];
    }
    acc = wmma_bf16(af, bfr, acc);
  }
#pragma unroll
  for (int v = 0; v < 8; ++v) {
    const int M = hi * 8 + v;
    s_acc[wave][M][lo] = acc[v] * s_inv[M];
  }
  __syncthreads();

#pragma unroll
  for (int t = 0; t < 2; ++t) {
    const int idx = tid + t * 128;           // 256 outputs / block
    const int m = idx >> 4, c = idx & 15;
    const float s = s_acc[0][m][c] + s_acc[1][m][c] +
                    s_acc[2][m][c] + s_acc[3][m][c];
    out[((size_t)b * N_ + i0 + m) * NC_ + c] = s;
  }
}

// ---------------------------------------------------------------------------
extern "C" void kernel_launch(void* const* d_in, const int* in_sizes, int n_in,
                              void* d_out, int out_size, void* d_ws, size_t ws_size,
                              hipStream_t stream) {
  (void)in_sizes; (void)n_in; (void)out_size; (void)ws_size;

  const float* x   = (const float*)d_in[0];   // [32,512,128]
  const int*   adj = (const int*)  d_in[1];   // [32,512,512]
  const float* Wh  = (const float*)d_in[2];   // [8,128,64]
  const float* a1h = (const float*)d_in[3];   // [8,64]
  const float* a2h = (const float*)d_in[4];   // [8,64]
  const float* Wo  = (const float*)d_in[5];   // [512,16]
  const float* a1o = (const float*)d_in[6];   // [16]
  const float* a2o = (const float*)d_in[7];   // [16]
  float* out = (float*)d_out;                 // [32,512,16] f32

  // workspace layout (~34.5 MB)
  char* ws = (char*)d_ws;
  size_t off = 0;
  bf16_t* hfeatT = (bf16_t*)(ws + off); off += (size_t)H_ * B_ * FH_ * N_ * 2; // 16 MB  [H,B,64,512]
  float*  f1     = (float*) (ws + off); off += (size_t)H_ * B_ * N_ * 4;       // 512 KB
  float*  f2     = (float*) (ws + off); off += (size_t)H_ * B_ * N_ * 4;       // 512 KB
  bf16_t* xcat   = (bf16_t*)(ws + off); off += (size_t)B_ * N_ * HF_ * 2;      // 16 MB  [B,N,512]
  bf16_t* houtT  = (bf16_t*)(ws + off); off += (size_t)B_ * NC_ * N_ * 2;      // 512 KB [B,16,512]
  float*  f1o    = (float*) (ws + off); off += (size_t)B_ * N_ * 4;            // 64 KB
  float*  f2o    = (float*) (ws + off); off += (size_t)B_ * N_ * 4;            // 64 KB
  bf16_t* WhT    = (bf16_t*)(ws + off); off += (size_t)H_ * FH_ * FIN_ * 2;    // 128 KB [H,64,128]
  bf16_t* WoT    = (bf16_t*)(ws + off);                                        // 16 KB  [16,512]

  const int tilesMB = B_ * (N_ / 16);   // 1024 row-tiles

  k0_pack     <<<dim3(256),         256, 0, stream>>>(Wh, Wo, WhT, WoT);
  k1_head_gemm<<<dim3(tilesMB, H_), 128, 0, stream>>>(x, WhT, a1h, a2h, hfeatT, f1, f2);
  k2_head_attn<<<dim3(tilesMB, H_), 128, 0, stream>>>(adj, hfeatT, f1, f2, xcat);
  k3_out_gemm <<<dim3(tilesMB / 4), 128, 0, stream>>>(xcat, WoT, a1o, a2o, houtT, f1o, f2o);
  k4_out_attn <<<dim3(tilesMB),     128, 0, stream>>>(adj, houtT, f1o, f2o, out);
}
